// HeteroGGNNConv_59854664237684
// MI455X (gfx1250) — matrix operand order repaired and checked by reading, not compiled, not run on hardware
//
#include <hip/hip_runtime.h>
#include <hip/hip_bf16.h>
#include <string.h>

// ---------------------------------------------------------------------------
// CDNA5 (gfx1250) HeteroGGNNConv: bf16 WMMA GEMMs + L2-resident scatter-sum
// ---------------------------------------------------------------------------

typedef __bf16 bf16_t;
typedef __attribute__((ext_vector_type(16))) __bf16 v16bf;
typedef __attribute__((ext_vector_type(8)))  float  v8f;

union FragBF {
    v16bf v;
    uint4 u[2];
};

__device__ __forceinline__ v8f wmma_bf16(const FragBF& a, const FragBF& b, v8f c) {
    return __builtin_amdgcn_wmma_f32_16x16x32_bf16(
        /*neg_a=*/false, a.v, /*neg_b=*/false, b.v,
        /*c_mod=*/(short)0, c, /*reuse_a=*/false, /*reuse_b=*/false);
}

// --------------------------- f32 -> bf16 convert ---------------------------
__global__ __launch_bounds__(256) void cvt_f32_bf16(const float* __restrict__ src,
                                                    bf16_t* __restrict__ dst, int n4) {
    int i = blockIdx.x * blockDim.x + threadIdx.x;
    if (i >= n4) return;
    float4 f = ((const float4*)src)[i];
    union { __bf16 h[4]; uint2 v; } p;
    p.h[0] = (__bf16)f.x; p.h[1] = (__bf16)f.y;
    p.h[2] = (__bf16)f.z; p.h[3] = (__bf16)f.w;
    ((uint2*)dst)[i] = p.v;
}

// ------------------- GEMM (NT): OUT[M,NOUT] = X[M,128] @ W[NOUT,128]^T -----
// One wave32 computes a 32x32 output tile = 2x2 WMMA tiles (register
// blocking: each A/B fragment feeds 2 WMMAs -> 8 b128 loads per 4 WMMAs).
// block = 4 waves -> 128 consecutive columns. grid = (ceil(M/32), NOUT/128).
// NOUT is compile-time so row addressing is shift+add (no v_mul_u64).
template <bool OUT_BF16, bool HAS_BIAS, int NOUT>
__global__ __launch_bounds__(128)
void gemm_nt_wmma(const bf16_t* __restrict__ X, const bf16_t* __restrict__ W,
                  const float* __restrict__ bias, void* __restrict__ outv,
                  int M) {
    const int lane = threadIdx.x & 31;
    const int wave = threadIdx.x >> 5;
    const int hb   = lane >> 4;      // half-wave: 0 or 1
    const int l16  = lane & 15;

    const int m0 = blockIdx.x * 32;          // first M-tile (always valid)
    const int m1 = m0 + 16;                  // second M-tile (may be tail)
    const int n0 = (blockIdx.y * 4 + wave) * 32;
    const int n1 = n0 + 16;

    // A rows this lane loads (clamped: tail tile reads stay in range)
    int r0 = m0 + l16;               r0 = r0 < M ? r0 : M - 1;
    int r1 = m1 + l16;               r1 = r1 < M ? r1 : M - 1;
    const bf16_t* xrow0 = X + (size_t)r0 * 128;
    const bf16_t* xrow1 = X + (size_t)r1 * 128;
    // W rows (== output columns) this lane owns
    const int n0c = n0 + l16;
    const int n1c = n1 + l16;
    const bf16_t* wrow0 = W + (size_t)n0c * 128;
    const bf16_t* wrow1 = W + (size_t)n1c * 128;

    v8f c00 = {0.f,0.f,0.f,0.f,0.f,0.f,0.f,0.f};
    v8f c01 = c00, c10 = c00, c11 = c00;

#pragma unroll
    for (int k0 = 0; k0 < 128; k0 += 32) {
        FragBF a0, a1, b0, b1;
        // A fragment (16x32 bf16): lanes 0-15: K=k0+[0..7],[16..23];
        // lanes 16-31: K=k0+[8..15],[24..31]   (ISA 7.12.2)
        a0.u[0] = *(const uint4*)(xrow0 + k0 + 8 * hb);
        a0.u[1] = *(const uint4*)(xrow0 + k0 + 16 + 8 * hb);
        a1.u[0] = *(const uint4*)(xrow1 + k0 + 8 * hb);
        a1.u[1] = *(const uint4*)(xrow1 + k0 + 16 + 8 * hb);
        // B fragment (32x16 bf16, B[k][n] = W[n][k]):
        // lane: column n, K = k0 + 8*hb + [0..7], k0 + 16 + 8*hb + [0..7] (same as A)
        b0.u[0] = *(const uint4*)(wrow0 + k0 + 8 * hb);
        b0.u[1] = *(const uint4*)(wrow0 + k0 + 16 + 8 * hb);
        b1.u[0] = *(const uint4*)(wrow1 + k0 + 8 * hb);
        b1.u[1] = *(const uint4*)(wrow1 + k0 + 16 + 8 * hb);
        c00 = wmma_bf16(a0, b0, c00);
        c01 = wmma_bf16(a0, b1, c01);
        c10 = wmma_bf16(a1, b0, c10);
        c11 = wmma_bf16(a1, b1, c11);
    }

    float bv0 = 0.f, bv1 = 0.f;
    if (HAS_BIAS) { bv0 = bias[n0c]; bv1 = bias[n1c]; }

    // C/D layout: VGPR r -> row (tile_base + r + 8*half), col = lane%16.
    // m0 tile is always fully valid -> unconditional straight-line stores.
#pragma unroll
    for (int r = 0; r < 8; ++r) {
        size_t row = (size_t)(m0 + r + 8 * hb) * NOUT;
        if (OUT_BF16) {
            ((bf16_t*)outv)[row + n0c] = (bf16_t)(c00[r] + bv0);
            ((bf16_t*)outv)[row + n1c] = (bf16_t)(c01[r] + bv1);
        } else {
            ((float*)outv)[row + n0c] = c00[r] + bv0;
            ((float*)outv)[row + n1c] = c01[r] + bv1;
        }
    }
    if (m1 < M) {   // wave-uniform scalar branch (tail tile only)
#pragma unroll
        for (int r = 0; r < 8; ++r) {
            size_t row = (size_t)(m1 + r + 8 * hb) * NOUT;
            if (OUT_BF16) {
                ((bf16_t*)outv)[row + n0c] = (bf16_t)(c10[r] + bv0);
                ((bf16_t*)outv)[row + n1c] = (bf16_t)(c11[r] + bv1);
            } else {
                ((float*)outv)[row + n0c] = c10[r] + bv0;
                ((float*)outv)[row + n1c] = c11[r] + bv1;
            }
        }
    }
}

// ------------------- scatter-sum: agg[dst] += sx_bf16[src] -----------------
// 32 lanes per edge, 4 channels per lane. sx (12.8MB bf16) and agg (25.6MB
// f32) are L2-resident on MI455X (192MB L2) -> atomics stay in L2.
__global__ __launch_bounds__(256)
void scatter_add(float* __restrict__ agg, const bf16_t* __restrict__ sx,
                 const int* __restrict__ edge, int E) {
    long gid = (long)blockIdx.x * blockDim.x + threadIdx.x;
    long e = gid >> 5;
    if (e >= E) return;
    int lane = (int)(gid & 31);
    if (lane == 0)  // keep the edge-index stream ahead of us in L2
        __builtin_prefetch(edge + e + 4096, 0, 1);
    int src = edge[e];
    int dst = edge[(size_t)E + e];
    union { uint2 u; __bf16 h[4]; } d;
    d.u = *(const uint2*)(sx + (size_t)src * 128 + lane * 4);
    float* ap = agg + (size_t)dst * 128 + lane * 4;
#pragma unroll
    for (int i = 0; i < 4; ++i)
        atomicAdd(ap + i, (float)d.h[i]);
}

// ------------------- GRU combine + merge epilogues -------------------------
// MODE 0: out = relu(gru)           (edge type e1 -> out_b)
// MODE 1: tmp = gru                 (edge type e2)
// MODE 2: out = relu((tmp+gru)/2)   (edge type e3 -> out_a)
__device__ __forceinline__ float fast_sigmoid(float x) {
    return 1.f / (1.f + __expf(-x));
}
__device__ __forceinline__ float fast_tanh(float x) {
    return 1.f - 2.f / (__expf(2.f * x) + 1.f);
}

template <int MODE>
__global__ __launch_bounds__(256)
void gru_combine(const float* __restrict__ gi, const float* __restrict__ gh,
                 const float* __restrict__ agg, float* __restrict__ tmp,
                 float* __restrict__ out, int N) {
    int idx = blockIdx.x * blockDim.x + threadIdx.x;
    if (idx >= N * 128) return;
    int nrow = idx >> 7, c = idx & 127;
    size_t g = (size_t)nrow * 384 + c;
    float ir = gi[g],       hr = gh[g];
    float iz = gi[g + 128], hz = gh[g + 128];
    float in = gi[g + 256], hn = gh[g + 256];
    float r  = fast_sigmoid(ir + hr);
    float z  = fast_sigmoid(iz + hz);
    float nn = fast_tanh(in + r * hn);
    float h  = agg[idx];
    float val = (1.f - z) * nn + z * h;
    if (MODE == 0)      out[idx] = fmaxf(val, 0.f);
    else if (MODE == 1) tmp[idx] = val;
    else                out[idx] = fmaxf((tmp[idx] + val) * 0.5f, 0.f);
}

// ---------------------------------------------------------------------------
extern "C" void kernel_launch(void* const* d_in, const int* in_sizes, int n_in,
                              void* d_out, int out_size, void* d_ws, size_t ws_size,
                              hipStream_t stream) {
    const int D = 128;
    const int N = in_sizes[0] / D;       // 50000
    const int E = in_sizes[2] / 2;       // 500000

    const float* x_a = (const float*)d_in[0];
    const float* x_b = (const float*)d_in[1];
    const int* edges[3] = {(const int*)d_in[2], (const int*)d_in[3], (const int*)d_in[4]};
    const float *Ws[3], *Wt[3], *Wih[3], *Whh[3], *Bih[3], *Bhh[3];
    for (int t = 0; t < 3; ++t) {
        int b = 5 + 6 * t;
        Ws[t]  = (const float*)d_in[b + 0];
        Wt[t]  = (const float*)d_in[b + 1];
        Wih[t] = (const float*)d_in[b + 2];
        Whh[t] = (const float*)d_in[b + 3];
        Bih[t] = (const float*)d_in[b + 4];
        Bhh[t] = (const float*)d_in[b + 5];
    }

    // ---- workspace carve (256B aligned) ----
    char* wsp = (char*)d_ws;
    auto alloc = [&](size_t bytes) -> char* {
        char* p = wsp;
        wsp += (bytes + 255) & ~(size_t)255;
        return p;
    };
    bf16_t* xa_bf  = (bf16_t*)alloc((size_t)N * D * 2);
    bf16_t* xb_bf  = (bf16_t*)alloc((size_t)N * D * 2);
    bf16_t *ws_bf[3], *wt_bf[3], *wih_bf[3], *whh_bf[3];
    for (int t = 0; t < 3; ++t) {
        ws_bf[t]  = (bf16_t*)alloc((size_t)D * D * 2);
        wt_bf[t]  = (bf16_t*)alloc((size_t)D * D * 2);
        wih_bf[t] = (bf16_t*)alloc((size_t)3 * D * D * 2);
        whh_bf[t] = (bf16_t*)alloc((size_t)3 * D * D * 2);
    }
    bf16_t* sx_bf  = (bf16_t*)alloc((size_t)N * D * 2);
    bf16_t* tx_bf  = (bf16_t*)alloc((size_t)N * D * 2);
    float*  agg    = (float*) alloc((size_t)N * D * 4);
    bf16_t* agg_bf = (bf16_t*)alloc((size_t)N * D * 2);
    float*  gi     = (float*) alloc((size_t)N * 3 * D * 4);
    float*  gh     = (float*) alloc((size_t)N * 3 * D * 4);
    float*  tmp    = (float*) alloc((size_t)N * D * 4);

    float* out_a = (float*)d_out;                 // [N, D]
    float* out_b = (float*)d_out + (size_t)N * D; // [N, D]

    auto cvt = [&](const float* s, bf16_t* d, int n) {
        int n4 = n / 4;
        cvt_f32_bf16<<<(n4 + 255) / 256, 256, 0, stream>>>(s, d, n4);
    };

    // ---- convert activations + weights to bf16 once ----
    cvt(x_a, xa_bf, N * D);
    cvt(x_b, xb_bf, N * D);
    for (int t = 0; t < 3; ++t) {
        cvt(Ws[t],  ws_bf[t],  D * D);
        cvt(Wt[t],  wt_bf[t],  D * D);
        cvt(Wih[t], wih_bf[t], 3 * D * D);
        cvt(Whh[t], whh_bf[t], 3 * D * D);
    }

    const bf16_t* src_bf[3] = {xa_bf, xb_bf, xb_bf};  // e1: a->b, e2/e3: b->a
    const bf16_t* tgt_bf[3] = {xb_bf, xa_bf, xa_bf};

    dim3 gBlk(128);
    dim3 gGridD((N + 31) / 32, 1);   // NOUT = 128 -> 128 cols per block
    dim3 gGrid3D((N + 31) / 32, 3);  // NOUT = 384
    int combBlocks = (N * D + 255) / 256;
    int scatBlocks = (int)(((long)E * 32 + 255) / 256);

    for (int t = 0; t < 3; ++t) {
        // sx = x_src @ Ws^T  (bf16 out: halves gather payload; L2-resident)
        gemm_nt_wmma<true, false, 128><<<gGridD, gBlk, 0, stream>>>(
            src_bf[t], ws_bf[t], nullptr, (void*)sx_bf, N);
        // tx = x_tgt @ Wt^T  (bf16 out: only feeds the gi GEMM)
        gemm_nt_wmma<true, false, 128><<<gGridD, gBlk, 0, stream>>>(
            tgt_bf[t], wt_bf[t], nullptr, (void*)tx_bf, N);
        // agg = segment_sum(sx[edge[0]], edge[1])  (f32 atomics in L2)
        hipMemsetAsync(agg, 0, (size_t)N * D * 4, stream);
        scatter_add<<<scatBlocks, 256, 0, stream>>>(agg, sx_bf, edges[t], E);
        cvt(agg, agg_bf, N * D);
        // gi = tx @ wih^T + bih ; gh = agg @ whh^T + bhh   (f32 out)
        gemm_nt_wmma<false, true, 384><<<gGrid3D, gBlk, 0, stream>>>(
            tx_bf, wih_bf[t], Bih[t], (void*)gi, N);
        gemm_nt_wmma<false, true, 384><<<gGrid3D, gBlk, 0, stream>>>(
            agg_bf, whh_bf[t], Bhh[t], (void*)gh, N);
        // fused GRU gates + merge epilogue
        if (t == 0)
            gru_combine<0><<<combBlocks, 256, 0, stream>>>(gi, gh, agg, tmp, out_b, N);
        else if (t == 1)
            gru_combine<1><<<combBlocks, 256, 0, stream>>>(gi, gh, agg, tmp, nullptr, N);
        else
            gru_combine<2><<<combBlocks, 256, 0, stream>>>(gi, gh, agg, tmp, out_a, N);
    }
    (void)n_in; (void)out_size; (void)ws_size;
}
